// Attention_50165218017281
// MI455X (gfx1250) — compile-verified, matrix operand order checked by reading, not run on hardware
//
#include <hip/hip_runtime.h>

// ---------------- problem constants (from reference) ----------------
#define B_   4
#define S_   1024
#define D_   4096
#define H_   32
#define KH_  8
#define HD_  128
#define REP_ 4
#define QKVN 6144            // H*HD + 2*KH*HD
#define BS_  (B_ * S_)       // 4096 rows
#define SCALE 0.08838834764831845f  // 1/sqrt(128)
#define KCH  64              // K elements per TDM chunk (128B per row)

typedef __attribute__((ext_vector_type(16))) __bf16 v16bf;
typedef __attribute__((ext_vector_type(8)))  float  v8f;
typedef __attribute__((ext_vector_type(4)))  unsigned v4u;
typedef __attribute__((ext_vector_type(8)))  int    v8i_;
typedef __attribute__((ext_vector_type(4)))  int    v4i_;

// ---------------- helpers ----------------
static __device__ __forceinline__ unsigned short f2bf(float x) {
  unsigned u = __float_as_uint(x);
  unsigned r = 0x7FFFu + ((u >> 16) & 1u);   // round-to-nearest-even
  return (unsigned short)((u + r) >> 16);
}

// 16-bit A-style fragment (ISA 7.12.2, 16x32): lane owns row (lane&15);
// dword index = K/2 -> {d, d+8} + 4g for d=0..3, g = lane>>4.
static __device__ __forceinline__ v16bf frag_k32(const unsigned short* rowbase,
                                                 int k0, unsigned g) {
  union { v16bf v; unsigned u[8]; } f;
  const unsigned* p = reinterpret_cast<const unsigned*>(rowbase + k0) + 4u * g;
#pragma unroll
  for (int d = 0; d < 4; ++d) {
    f.u[d]     = p[d];
    f.u[d + 4] = p[d + 8];
  }
  return f.v;
}

static __device__ __forceinline__ v8f wmma_bf16(v16bf a, v16bf b, v8f c) {
  return __builtin_amdgcn_wmma_f32_16x16x32_bf16(false, a, false, b,
                                                 (short)0, c, false, false);
}

// Issue TDM: load a [64 rows x KCH] bf16 tile (row stride = K elements) from
// global into LDS.  D# per ISA 08_async_tensor §8 (2-D tile, groups 2/3 = 0).
// 6-arg builtin form (clang-23 / therock-10.0 headers).
static __device__ __forceinline__ void tdm_load_tile(const unsigned short* gsrc,
                                                     unsigned lds_byte_addr,
                                                     int K, int M) {
  unsigned long long ga = (unsigned long long)(size_t)gsrc;
  v4u g0;
  g0[0] = 1u;                                   // count=1, user descriptor
  g0[1] = lds_byte_addr;                        // lds_addr [63:32]
  g0[2] = (unsigned)(ga & 0xFFFFFFFFull);       // global_addr lo
  g0[3] = (unsigned)((ga >> 32) & 0x01FFFFFFull) | 0x80000000u; // addr hi | type=2
  v8i_ g1;
  g1[0] = (1 << 16);                            // wg_mask=0, data_size=1 (2 bytes)
  g1[1] = (int)((K & 0xFFFF) << 16);            // tensor_dim0[15:0] at bits 63:48
  g1[2] = (int)(((unsigned)K >> 16) | ((unsigned)(M & 0xFFFF) << 16)); // dim0 hi | dim1 lo
  g1[3] = (int)(((unsigned)M >> 16) | ((unsigned)KCH << 16));          // dim1 hi | tile_dim0
  g1[4] = 64;                                   // tile_dim1=64 rows, tile_dim2=0
  g1[5] = K;                                    // tensor_dim0_stride[31:0]
  g1[6] = 0;                                    // stride hi, dim1_stride lo
  g1[7] = 0;
  v4i_ z4 = {0, 0, 0, 0};
  v8i_ z8 = {0, 0, 0, 0, 0, 0, 0, 0};
  __builtin_amdgcn_tensor_load_to_lds(g0, g1, z4, z4, z8, 0);
}

// ---------------- stage 1: casts / transposes ----------------
__global__ __launch_bounds__(256) void cast_bf16_kernel(const float* __restrict__ src,
                                                        unsigned short* __restrict__ dst,
                                                        size_t n) {
  size_t i = (size_t)blockIdx.x * blockDim.x + threadIdx.x;
  if (i < n) dst[i] = f2bf(src[i]);
}

// src: [K, N] f32 row-major  ->  dst: [N, K] bf16 row-major (B-transposed)
__global__ __launch_bounds__(256) void transpose_cast_kernel(const float* __restrict__ src,
                                                             unsigned short* __restrict__ dst,
                                                             int K, int N) {
  size_t i = (size_t)blockIdx.x * blockDim.x + threadIdx.x;
  if (i >= (size_t)K * N) return;
  int k = (int)(i / N);
  int n = (int)(i % N);
  dst[(size_t)n * K + k] = f2bf(src[i]);
}

// ---------------- stage 2/5: bf16 WMMA GEMM with TDM-staged A tile ---------
// A: [M,K] bf16 row-major; BT: [N,K] bf16 row-major; C: [M,N] f32.
// Block: 256 threads = 8 waves (2x4); wave computes 32x64; block tile 64x256.
// A 64xKCH tile double-buffered in LDS via tensor_load_to_lds (wave 0 issues).
__global__ __launch_bounds__(256) void gemm_bf16_nt(const unsigned short* __restrict__ A,
                                                    const unsigned short* __restrict__ BT,
                                                    float* __restrict__ C,
                                                    int M, int N, int K) {
  __shared__ unsigned short tileA[2][64 * KCH];   // 2 x 8 KB

  const unsigned lane = threadIdx.x & 31u;
  const unsigned wave = threadIdx.x >> 5;
  const unsigned wm = wave >> 2;          // 0..1  (M)
  const unsigned wn = wave & 3u;          // 0..3  (N)
  const unsigned g   = lane >> 4;
  const unsigned l15 = lane & 15u;
  const int m0 = blockIdx.x * 64;                     // block row base
  const int n0 = blockIdx.y * 256 + (int)wn * 64;     // wave col base
  const int mw = (int)wm * 32;                        // wave row offset in tile

  const unsigned short* bptr[4];
#pragma unroll
  for (int j = 0; j < 4; ++j)
    bptr[j] = BT + (size_t)(n0 + j * 16 + (int)l15) * K;

  v8f acc[2][4];
#pragma unroll
  for (int i = 0; i < 2; ++i)
#pragma unroll
    for (int j = 0; j < 4; ++j) acc[i][j] = (v8f){0, 0, 0, 0, 0, 0, 0, 0};

  const int nch = K / KCH;
  const unsigned lds0 = (unsigned)(size_t)&tileA[0][0];
  const unsigned lds1 = (unsigned)(size_t)&tileA[1][0];

  if (wave == 0) {
    tdm_load_tile(A + (size_t)m0 * K, lds0, K, M);
    __builtin_amdgcn_s_wait_tensorcnt(0);
  }
  __syncthreads();

  for (int kc = 0; kc < nch; ++kc) {
    const int buf = kc & 1;
    if (wave == 0 && kc + 1 < nch) {
      tdm_load_tile(A + (size_t)m0 * K + (size_t)(kc + 1) * KCH,
                    buf ? lds0 : lds1, K, M);
    }
    // prefetch next B chunk (streams through L2)
    if (kc + 1 < nch) {
#pragma unroll
      for (int j = 0; j < 4; ++j)
        __builtin_prefetch(bptr[j] + (size_t)(kc + 1) * KCH, 0, 1);
    }

    const unsigned short* a0 = &tileA[buf][(mw + (int)l15) * KCH];
    const unsigned short* a1 = a0 + 16 * KCH;
#pragma unroll
    for (int ks = 0; ks < KCH; ks += 32) {
      v16bf af0 = frag_k32(a0, ks, g);
      v16bf af1 = frag_k32(a1, ks, g);
      v16bf bf[4];
#pragma unroll
      for (int j = 0; j < 4; ++j)
        bf[j] = frag_k32(bptr[j], kc * KCH + ks, g);
#pragma unroll
      for (int j = 0; j < 4; ++j) {
        acc[0][j] = wmma_bf16(af0, bf[j], acc[0][j]);
        acc[1][j] = wmma_bf16(af1, bf[j], acc[1][j]);
      }
    }

    if (wave == 0) __builtin_amdgcn_s_wait_tensorcnt(0);
    __syncthreads();
  }

  // D fragment: element (M = r + 8g, N = lane&15) in component r.
#pragma unroll
  for (int i = 0; i < 2; ++i) {
#pragma unroll
    for (int j = 0; j < 4; ++j) {
      union { v8f v; float f[8]; } u;
      u.v = acc[i][j];
#pragma unroll
      for (int r = 0; r < 8; ++r) {
        C[(size_t)(m0 + mw + i * 16 + r + 8 * (int)g) * N + n0 + j * 16 + (int)l15] = u.f[r];
      }
    }
  }
}

// ---------------- stage 3: RoPE + layout shuffles ----------------
__global__ __launch_bounds__(256) void rope_q_kernel(const float* __restrict__ qkv,
                                                     const float* __restrict__ freqs,
                                                     unsigned short* __restrict__ qb) {
  size_t i = (size_t)blockIdx.x * blockDim.x + threadIdx.x;
  const size_t total = (size_t)B_ * S_ * H_ * (HD_ / 2);
  if (i >= total) return;
  int p = (int)(i % (HD_ / 2)); size_t t = i / (HD_ / 2);
  int h = (int)(t % H_); t /= H_;
  int s = (int)(t % S_); int b = (int)(t / S_);
  const float* row = qkv + ((size_t)b * S_ + s) * QKVN + (size_t)h * HD_;
  float a = row[2 * p], bv = row[2 * p + 1];
  float sn, cs;
  __sincosf(freqs[(size_t)s * (HD_ / 2) + p], &sn, &cs);
  unsigned short* o = qb + (((size_t)(b * H_ + h) * S_) + s) * HD_ + 2 * p;
  o[0] = f2bf(a * cs - bv * sn);
  o[1] = f2bf(a * sn + bv * cs);
}

__global__ __launch_bounds__(256) void rope_k_kernel(const float* __restrict__ qkv,
                                                     const float* __restrict__ freqs,
                                                     unsigned short* __restrict__ kb) {
  size_t i = (size_t)blockIdx.x * blockDim.x + threadIdx.x;
  const size_t total = (size_t)B_ * S_ * KH_ * (HD_ / 2);
  if (i >= total) return;
  int p = (int)(i % (HD_ / 2)); size_t t = i / (HD_ / 2);
  int kh = (int)(t % KH_); t /= KH_;
  int s = (int)(t % S_); int b = (int)(t / S_);
  const float* row = qkv + ((size_t)b * S_ + s) * QKVN + (size_t)(H_ * HD_) + (size_t)kh * HD_;
  float a = row[2 * p], bv = row[2 * p + 1];
  float sn, cs;
  __sincosf(freqs[(size_t)s * (HD_ / 2) + p], &sn, &cs);
  unsigned short* o = kb + (((size_t)(b * KH_ + kh) * S_) + s) * HD_ + 2 * p;
  o[0] = f2bf(a * cs - bv * sn);
  o[1] = f2bf(a * sn + bv * cs);
}

// v -> transposed [b, kh, HD, S] so P@V B-fragments are contiguous in keys
__global__ __launch_bounds__(256) void conv_v_kernel(const float* __restrict__ qkv,
                                                     unsigned short* __restrict__ vt) {
  size_t i = (size_t)blockIdx.x * blockDim.x + threadIdx.x;
  const size_t total = (size_t)B_ * S_ * KH_ * HD_;
  if (i >= total) return;
  int hd = (int)(i % HD_); size_t t = i / HD_;
  int kh = (int)(t % KH_); t /= KH_;
  int s = (int)(t % S_); int b = (int)(t / S_);
  float v = qkv[((size_t)b * S_ + s) * QKVN + (size_t)(H_ * HD_ + KH_ * HD_) + (size_t)kh * HD_ + hd];
  vt[(((size_t)(b * KH_ + kh) * HD_) + hd) * S_ + s] = f2bf(v);
}

// ---------------- stage 4: causal flash attention (WMMA bf16) ----------------
// grid = (S/64, H, B), block = 128 (4 waves), wave owns 16 q rows.
__global__ __launch_bounds__(128) void flash_attn_kernel(const unsigned short* __restrict__ Q,
                                                         const unsigned short* __restrict__ Kb,
                                                         const unsigned short* __restrict__ VT,
                                                         unsigned short* __restrict__ Oout) {
  __shared__ unsigned short plds[4 * 16 * 32];  // per-wave 16x32 P tile (bf16)

  const unsigned lane = threadIdx.x & 31u;
  const unsigned wave = threadIdx.x >> 5;
  const unsigned g   = lane >> 4;
  const unsigned l15 = lane & 15u;
  const int b = blockIdx.z, h = blockIdx.y;
  const int kvh = h / REP_;
  const int q0 = blockIdx.x * 64 + (int)wave * 16;

  const unsigned short* qbase = Q  + (((size_t)(b * H_ + h) * S_) + q0 + (int)l15) * HD_;
  const unsigned short* kbase = Kb + ((size_t)(b * KH_ + kvh) * S_) * HD_;
  const unsigned short* vtb   = VT + ((size_t)(b * KH_ + kvh) * HD_) * S_;
  unsigned short* myp = &plds[wave * 512];

  v16bf qf[4];
#pragma unroll
  for (int c = 0; c < 4; ++c) qf[c] = frag_k32(qbase, 32 * c, g);

  v8f O[8];
  float m_i[8], l_i[8];
#pragma unroll
  for (int nt = 0; nt < 8; ++nt) O[nt] = (v8f){0,0,0,0,0,0,0,0};
#pragma unroll
  for (int r = 0; r < 8; ++r) { m_i[r] = -1e30f; l_i[r] = 0.f; }

  for (int k0 = 0; k0 <= q0 + 15; k0 += 32) {       // causal: stop at diagonal
    v8f s0 = {0,0,0,0,0,0,0,0}, s1 = s0;
#pragma unroll
    for (int c = 0; c < 4; ++c) {                   // K reduction over HD=128
      v16bf kf0 = frag_k32(kbase + (size_t)(k0 + (int)l15) * HD_,      32 * c, g);
      v16bf kf1 = frag_k32(kbase + (size_t)(k0 + 16 + (int)l15) * HD_, 32 * c, g);
      s0 = wmma_bf16(qf[c], kf0, s0);
      s1 = wmma_bf16(qf[c], kf1, s1);
    }

    union { v8f v; float f[8]; } u0, u1;
    u0.v = s0; u1.v = s1;
#pragma unroll
    for (int r = 0; r < 8; ++r) {
      const int qrow = q0 + r + 8 * (int)g;
      float a0 = u0.f[r] * SCALE + ((k0 + (int)l15)      <= qrow ? 0.f : -1e9f);
      float a1 = u1.f[r] * SCALE + ((k0 + 16 + (int)l15) <= qrow ? 0.f : -1e9f);
      float mr = fmaxf(a0, a1);
#pragma unroll
      for (int off = 1; off < 16; off <<= 1) mr = fmaxf(mr, __shfl_xor(mr, off, 32));
      float mnew  = fmaxf(m_i[r], mr);
      float alpha = __expf(m_i[r] - mnew);
      float p0 = __expf(a0 - mnew), p1 = __expf(a1 - mnew);
      float srow = p0 + p1;
#pragma unroll
      for (int off = 1; off < 16; off <<= 1) srow += __shfl_xor(srow, off, 32);
      l_i[r] = l_i[r] * alpha + srow;
      m_i[r] = mnew;
#pragma unroll
      for (int nt = 0; nt < 8; ++nt) O[nt][r] *= alpha;   // rescale accumulator
      // D-layout (row r+8g, col l15) -> LDS row-major P tile
      myp[(r + 8 * (int)g) * 32 + (int)l15]      = f2bf(p0);
      myp[(r + 8 * (int)g) * 32 + 16 + (int)l15] = f2bf(p1);
    }

    // read P back in A-fragment layout (same-wave DS ordering)
    v16bf pf = frag_k32(myp + (size_t)l15 * 32, 0, g);
#pragma unroll
    for (int nt = 0; nt < 8; ++nt) {
      v16bf vf = frag_k32(vtb + (size_t)(nt * 16 + (int)l15) * S_, k0, g);
      O[nt] = wmma_bf16(pf, vf, O[nt]);
    }
  }

  // epilogue: O / l, write bf16 into [B*S, H*HD]
#pragma unroll
  for (int nt = 0; nt < 8; ++nt) {
#pragma unroll
    for (int r = 0; r < 8; ++r) {
      float o = O[nt][r] / l_i[r];
      size_t row = (size_t)b * S_ + q0 + r + 8 * (int)g;
      Oout[row * (H_ * HD_) + (size_t)h * HD_ + nt * 16 + (int)l15] = f2bf(o);
    }
  }
}

// ---------------- launch ----------------
extern "C" void kernel_launch(void* const* d_in, const int* in_sizes, int n_in,
                              void* d_out, int out_size, void* d_ws, size_t ws_size,
                              hipStream_t stream) {
  const float* x     = (const float*)d_in[0];
  const float* freqs = (const float*)d_in[1];
  // d_in[2] = additive mask: causal mask is computed in-kernel.
  const float* wq    = (const float*)d_in[3];
  const float* wk    = (const float*)d_in[4];
  const float* wv    = (const float*)d_in[5];
  const float* wo    = (const float*)d_in[6];
  float* out = (float*)d_out;

  char* ws = (char*)d_ws;
  size_t off = 0;
  auto alloc = [&](size_t bytes) -> void* {
    void* p = ws + off;
    off += (bytes + 255) & ~(size_t)255;
    return p;
  };

  unsigned short* xb    = (unsigned short*)alloc((size_t)BS_ * D_ * 2);
  unsigned short* wqkvT = (unsigned short*)alloc((size_t)QKVN * D_ * 2);
  unsigned short* woT   = (unsigned short*)alloc((size_t)D_ * (H_ * HD_) * 2);
  float*          qkv   = (float*)alloc((size_t)BS_ * QKVN * 4);
  unsigned short* qb    = (unsigned short*)alloc((size_t)B_ * H_ * S_ * HD_ * 2);
  unsigned short* kb    = (unsigned short*)alloc((size_t)B_ * KH_ * S_ * HD_ * 2);
  unsigned short* vt    = (unsigned short*)alloc((size_t)B_ * KH_ * HD_ * S_ * 2);
  unsigned short* attn  = (unsigned short*)alloc((size_t)BS_ * (H_ * HD_) * 2);

  // stage 1: casts + weight transposes (fused QKV B^T)
  {
    size_t n = (size_t)BS_ * D_;
    cast_bf16_kernel<<<dim3((unsigned)((n + 255) / 256)), 256, 0, stream>>>(x, xb, n);
  }
  transpose_cast_kernel<<<dim3((unsigned)(((size_t)D_ * (H_ * HD_) + 255) / 256)), 256, 0, stream>>>(
      wq, wqkvT, D_, H_ * HD_);
  transpose_cast_kernel<<<dim3((unsigned)(((size_t)D_ * (KH_ * HD_) + 255) / 256)), 256, 0, stream>>>(
      wk, wqkvT + (size_t)(H_ * HD_) * D_, D_, KH_ * HD_);
  transpose_cast_kernel<<<dim3((unsigned)(((size_t)D_ * (KH_ * HD_) + 255) / 256)), 256, 0, stream>>>(
      wv, wqkvT + (size_t)(H_ * HD_ + KH_ * HD_) * D_, D_, KH_ * HD_);
  transpose_cast_kernel<<<dim3((unsigned)(((size_t)(H_ * HD_) * D_ + 255) / 256)), 256, 0, stream>>>(
      wo, woT, H_ * HD_, D_);

  // stage 2: fused QKV projection (M=4096, N=6144, K=4096)
  gemm_bf16_nt<<<dim3(BS_ / 64, QKVN / 256), 256, 0, stream>>>(xb, wqkvT, qkv, BS_, QKVN, D_);

  // stage 3: RoPE + layout
  rope_q_kernel<<<dim3((unsigned)(((size_t)B_ * S_ * H_ * (HD_ / 2) + 255) / 256)), 256, 0, stream>>>(qkv, freqs, qb);
  rope_k_kernel<<<dim3((unsigned)(((size_t)B_ * S_ * KH_ * (HD_ / 2) + 255) / 256)), 256, 0, stream>>>(qkv, freqs, kb);
  conv_v_kernel<<<dim3((unsigned)(((size_t)B_ * S_ * KH_ * HD_ + 255) / 256)), 256, 0, stream>>>(qkv, vt);

  // stage 4: causal flash attention
  flash_attn_kernel<<<dim3(S_ / 64, H_, B_), 128, 0, stream>>>(qb, kb, vt, attn);

  // stage 5: output projection (M=4096, N=4096, K=4096) -> f32 out
  gemm_bf16_nt<<<dim3(BS_ / 64, D_ / 256), 256, 0, stream>>>(attn, woT, out, BS_, D_, D_);
}
